// GPTModel_33389075759110
// MI455X (gfx1250) — compile-verified, matrix operand order checked by reading, not run on hardware
//
#include <hip/hip_runtime.h>

// ---------------------------------------------------------------------------
// GPT forward for MI455X (gfx1250, wave32, WMMA bf16 16x16x32, fp32 accum).
// All GEMM operands bf16; weights pre-transposed to [N][K] so every WMMA
// fragment load is two unconditional 16-byte vector loads (no exec masking).
// Residual stream fp32. Workspace ~266 MB.
// ---------------------------------------------------------------------------

#define VOCAB 50257
#define DMODEL 384
#define SEQ 300
#define BATCH 16
#define NHEAD 16
#define DHEAD 24
#define NLAYER 8
#define DFF 1536
#define BT (BATCH * SEQ)   // 4800
#define TP 320             // padded attention row stride (10 * 32)

typedef __attribute__((ext_vector_type(16))) __bf16 v16bf;
typedef __attribute__((ext_vector_type(8)))  float  v8f;

union Frag {
  v16bf v;
  uint4 q[2];
};

// ---- WMMA 16x16x32 bf16 fragment load -------------------------------------
// Both A (row-major [16 x 32]) and B^T (row-major [16 x 32], lane = out col)
// use the same pattern: lane r = lane&15 picks the row, kg = (lane>>4)*8 the
// K sub-group; elems 0..7 = K kg..kg+7, elems 8..15 = K kg+16..kg+23.
// Caller guarantees in-bounds (padded buffers) and 16B alignment.
__device__ __forceinline__ v16bf load_frag(const __bf16* __restrict__ p, int ld) {
  int lane = threadIdx.x & 31;
  int r = lane & 15;
  int kg = (lane >> 4) << 3;
  const __bf16* row = p + (size_t)r * ld + kg;
  Frag f;
  f.q[0] = *(const uint4*)(row);
  f.q[1] = *(const uint4*)(row + 16);
  return f.v;
}

// Variant with K clamped to 24 (attention head dim): lanes 16..31 would carry
// K=24..31 in their second chunk -> zero it (register select, no branches).
__device__ __forceinline__ v16bf load_frag_k24(const __bf16* __restrict__ p, int ld) {
  int lane = threadIdx.x & 31;
  int r = lane & 15;
  int kg = (lane >> 4) << 3;
  const __bf16* row = p + (size_t)r * ld + kg;
  Frag f;
  f.q[0] = *(const uint4*)(row);
  uint4 hi = *(const uint4*)(row + 16);
  bool ok = lane < 16;
  f.q[1].x = ok ? hi.x : 0u;
  f.q[1].y = ok ? hi.y : 0u;
  f.q[1].z = ok ? hi.z : 0u;
  f.q[1].w = ok ? hi.w : 0u;
  return f.v;
}

#define WMMA_BF16(a, b, c) \
  __builtin_amdgcn_wmma_f32_16x16x32_bf16(false, (a), false, (b), (short)0, (c), false, false)

// ---------------- weight conversion kernels --------------------------------
// Generic batched transpose+convert: in[l][r][c] (f32) -> out[l][c][r] (bf16).
__global__ void conv_transpose_kernel(const float* __restrict__ in,
                                      __bf16* __restrict__ out,
                                      int R, int C, int total) {
  int i = blockIdx.x * blockDim.x + threadIdx.x;
  if (i >= total) return;
  int c = i % C;
  int r = (i / C) % R;
  int l = i / (C * R);
  out[((size_t)l * C + c) * R + r] = (__bf16)in[i];
}

// Wq/Wk/Wv: [L][H][D][DH] (f32) -> bf16 [L][N=H*DH][K=D] (transposed weight)
__global__ void conv_qkv_kernel(const float* __restrict__ W,
                                __bf16* __restrict__ Wt, int total) {
  int i = blockIdx.x * blockDim.x + threadIdx.x;
  if (i >= total) return;
  int e = i % DHEAD;
  int d = (i / DHEAD) % DMODEL;
  int h = (i / (DHEAD * DMODEL)) % NHEAD;
  int l = i / (DHEAD * DMODEL * NHEAD);
  Wt[((size_t)l * DMODEL + (h * DHEAD + e)) * DMODEL + d] = (__bf16)W[i];
}

// V: [B*T][D] bf16 -> vT[b][col][TP] bf16 (per-batch transpose, padded stride)
__global__ void transpose_v_kernel(const __bf16* __restrict__ vb,
                                   __bf16* __restrict__ vT) {
  int i = blockIdx.x * blockDim.x + threadIdx.x;
  if (i >= BT * DMODEL) return;
  int col = i % DMODEL;
  int bt = i / DMODEL;
  int b = bt / SEQ, t = bt % SEQ;
  vT[((size_t)b * DMODEL + col) * TP + t] = vb[i];
}

// ---------------- embedding ------------------------------------------------
__global__ void embed_kernel(const int* __restrict__ idx,
                             const float* __restrict__ tok,
                             const float* __restrict__ pos,
                             float* __restrict__ x, __bf16* __restrict__ xb) {
  int bt = blockIdx.x;
  int d = threadIdx.x;  // 384
  int t = bt % SEQ;
  float v = tok[(size_t)idx[bt] * DMODEL + d] + pos[(size_t)t * DMODEL + d];
  x[(size_t)bt * DMODEL + d] = v;
  xb[(size_t)bt * DMODEL + d] = (__bf16)v;
}

// ---------------- generic bf16 WMMA GEMM (B pre-transposed) ----------------
// C[M,N] = A[M,K] @ Bt[N,K]^T + bias ; optional relu ; f32 and/or bf16 out.
// 128 threads = 4 waves; wave w owns a 16x64 strip (4 accumulators).
// M % 16 == 0, K % 32 == 0 guaranteed; only the N edge needs store guards.
__global__ void gemm_bf16_kernel(const __bf16* __restrict__ A,
                                 const __bf16* __restrict__ Bt,
                                 const float* __restrict__ bias,
                                 float* __restrict__ Cf, __bf16* __restrict__ Cb,
                                 int M, int N, int K,
                                 int lda, int ldb, int ldc, int relu) {
  int wave = threadIdx.x >> 5;
  int lane = threadIdx.x & 31;
  int m0 = blockIdx.x * 64 + wave * 16;
  if (m0 >= M) return;
  int nb = blockIdx.y * 64;
  const __bf16* Ap = A + (size_t)m0 * lda;
  const __bf16* Bp = Bt + (size_t)nb * ldb;

  v8f acc0 = {}, acc1 = {}, acc2 = {}, acc3 = {};
  for (int k0 = 0; k0 < K; k0 += 32) {
    v16bf a = load_frag(Ap + k0, lda);
    v16bf b0 = load_frag(Bp + k0, ldb);
    v16bf b1 = load_frag(Bp + (size_t)16 * ldb + k0, ldb);
    v16bf b2 = load_frag(Bp + (size_t)32 * ldb + k0, ldb);
    v16bf b3 = load_frag(Bp + (size_t)48 * ldb + k0, ldb);
    acc0 = WMMA_BF16(a, b0, acc0);
    acc1 = WMMA_BF16(a, b1, acc1);
    acc2 = WMMA_BF16(a, b2, acc2);
    acc3 = WMMA_BF16(a, b3, acc3);
  }

  int n = lane & 15;
  int mrow = m0 + ((lane >> 4) << 3);
  v8f accs[4] = {acc0, acc1, acc2, acc3};
#pragma unroll
  for (int j = 0; j < 4; ++j) {
    int gn = nb + j * 16 + n;
    if (gn >= N) continue;             // vocab edge only (per-lane, stores only)
    float bv = bias ? bias[gn] : 0.0f;
#pragma unroll
    for (int r = 0; r < 8; ++r) {
      int gm = mrow + r;
      float v = accs[j][r] + bv;
      if (relu) v = fmaxf(v, 0.0f);
      if (Cf) Cf[(size_t)gm * ldc + gn] = v;
      if (Cb) Cb[(size_t)gm * ldc + gn] = (__bf16)v;
    }
  }
}

// ---------------- attention: S = scale * Q @ K^T ---------------------------
// grid = (19 m-tiles, B*H); 32 threads. Causal: s-tiles 0..mt.
// A = Q rows (K clamped to 24); B = K rows (lane = key index, contiguous dh).
__global__ void attn_scores_kernel(const __bf16* __restrict__ qb,
                                   const __bf16* __restrict__ kb,
                                   float* __restrict__ S) {
  int mt = blockIdx.x;
  int bh = blockIdx.y;
  int b = bh >> 4, h = bh & 15;
  int lane = threadIdx.x & 31;
  int m0 = mt * 16;
  const float scale = 0.20412414523193154f;  // 1/sqrt(24)
  v16bf a = load_frag_k24(qb + (size_t)(b * SEQ + m0) * DMODEL + h * DHEAD, DMODEL);
  for (int st = 0; st <= mt; ++st) {
    int s0 = st * 16;
    v16bf bm = load_frag(kb + (size_t)(b * SEQ + s0) * DMODEL + h * DHEAD, DMODEL);
    v8f acc = {};
    acc = WMMA_BF16(a, bm, acc);
    int n = lane & 15;
    int s = s0 + n;
#pragma unroll
    for (int r = 0; r < 8; ++r) {
      int t = m0 + r + ((lane >> 4) << 3);
      if (t < SEQ) S[((size_t)bh * SEQ + t) * TP + s] = acc[r] * scale;
    }
  }
}

// ---------------- causal softmax -> bf16 P (zero-padded to TP) -------------
__global__ void softmax_kernel(const float* __restrict__ S, __bf16* __restrict__ P) {
  int t = blockIdx.x;
  int bh = blockIdx.y;
  int tid = threadIdx.x;  // 128
  const float* row = S + ((size_t)bh * SEQ + t) * TP;
  __bf16* prow = P + ((size_t)bh * SEQ + t) * TP;
  int len = t + 1;
  __shared__ float sm[128];

  float mx = -1e30f;
  for (int s = tid; s < len; s += 128) mx = fmaxf(mx, row[s]);
  sm[tid] = mx; __syncthreads();
  for (int o = 64; o > 0; o >>= 1) {
    if (tid < o) sm[tid] = fmaxf(sm[tid], sm[tid + o]);
    __syncthreads();
  }
  mx = sm[0]; __syncthreads();

  float sum = 0.0f;
  for (int s = tid; s < len; s += 128) sum += __expf(row[s] - mx);
  sm[tid] = sum; __syncthreads();
  for (int o = 64; o > 0; o >>= 1) {
    if (tid < o) sm[tid] += sm[tid + o];
    __syncthreads();
  }
  float inv = 1.0f / sm[0];

  for (int s = tid; s < TP; s += 128) {
    float p = (s < len) ? __expf(row[s] - mx) * inv : 0.0f;
    prow[s] = (__bf16)p;
  }
}

// ---------------- attention: O = P @ V -------------------------------------
// B = vT[b][dh][t] (padded stride TP): contiguous per lane. Columns dh>=24
// are computed on garbage rows but never stored.
__global__ void attn_av_kernel(const __bf16* __restrict__ P,
                               const __bf16* __restrict__ vT,
                               __bf16* __restrict__ ob) {
  int mt = blockIdx.x;
  int bh = blockIdx.y;
  int b = bh >> 4, h = bh & 15;
  int lane = threadIdx.x & 31;
  int m0 = mt * 16;
  int kend = m0 + 16; if (kend > SEQ) kend = SEQ;  // causal K extent
  const __bf16* Pp = P + ((size_t)bh * SEQ + m0) * TP;
  const __bf16* Vp0 = vT + ((size_t)b * DMODEL + h * DHEAD) * TP;
  const __bf16* Vp1 = Vp0 + (size_t)16 * TP;

  v8f acc0 = {}, acc1 = {};
  for (int k0 = 0; k0 < kend; k0 += 32) {
    v16bf a = load_frag(Pp + k0, TP);
    v16bf b0 = load_frag(Vp0 + k0, TP);
    v16bf b1 = load_frag(Vp1 + k0, TP);
    acc0 = WMMA_BF16(a, b0, acc0);
    acc1 = WMMA_BF16(a, b1, acc1);
  }

  int n = lane & 15;
  int mrow = m0 + ((lane >> 4) << 3);
  v8f accs[2] = {acc0, acc1};
#pragma unroll
  for (int j = 0; j < 2; ++j) {
    int dh = j * 16 + n;
    if (dh >= DHEAD) continue;
#pragma unroll
    for (int r = 0; r < 8; ++r) {
      int t = mrow + r;
      if (t < SEQ)
        ob[(size_t)(b * SEQ + t) * DMODEL + h * DHEAD + dh] = (__bf16)accs[j][r];
    }
  }
}

// ---------------- residual add + LayerNorm ---------------------------------
__global__ void ln_kernel(float* __restrict__ x, const float* __restrict__ y,
                          const float* __restrict__ g, const float* __restrict__ b,
                          __bf16* __restrict__ xb) {
  int row = blockIdx.x;
  int tid = threadIdx.x;  // 128, 3 elems each
  float* xr = x + (size_t)row * DMODEL;
  const float* yr = y ? y + (size_t)row * DMODEL : nullptr;
  __shared__ float sm[128];

  float local[3];
  float s = 0.0f;
#pragma unroll
  for (int j = 0; j < 3; ++j) {
    int d = tid + j * 128;
    float v = xr[d] + (yr ? yr[d] : 0.0f);
    local[j] = v;
    s += v;
  }
  sm[tid] = s; __syncthreads();
  for (int o = 64; o > 0; o >>= 1) {
    if (tid < o) sm[tid] += sm[tid + o];
    __syncthreads();
  }
  float mean = sm[0] * (1.0f / DMODEL);
  __syncthreads();

  float vs = 0.0f;
#pragma unroll
  for (int j = 0; j < 3; ++j) { float dv = local[j] - mean; vs += dv * dv; }
  sm[tid] = vs; __syncthreads();
  for (int o = 64; o > 0; o >>= 1) {
    if (tid < o) sm[tid] += sm[tid + o];
    __syncthreads();
  }
  float rstd = rsqrtf(sm[0] * (1.0f / DMODEL) + 1e-5f);

#pragma unroll
  for (int j = 0; j < 3; ++j) {
    int d = tid + j * 128;
    float o = (local[j] - mean) * rstd * g[d] + b[d];
    xr[d] = o;
    xb[(size_t)row * DMODEL + d] = (__bf16)o;
  }
}

// ---------------- loss -----------------------------------------------------
__global__ void zero_loss_kernel(float* __restrict__ loss) { *loss = 0.0f; }

__global__ void loss_kernel(const float* __restrict__ logits,
                            const int* __restrict__ targets,
                            float* __restrict__ loss) {
  int row = blockIdx.x;
  int tid = threadIdx.x;  // 256
  const float* lr = logits + (size_t)row * VOCAB;
  __shared__ float sm[256];

  float mx = -1e30f;
  for (int i = tid; i < VOCAB; i += 256) mx = fmaxf(mx, lr[i]);
  sm[tid] = mx; __syncthreads();
  for (int o = 128; o > 0; o >>= 1) {
    if (tid < o) sm[tid] = fmaxf(sm[tid], sm[tid + o]);
    __syncthreads();
  }
  mx = sm[0]; __syncthreads();

  float sum = 0.0f;
  for (int i = tid; i < VOCAB; i += 256) sum += __expf(lr[i] - mx);
  sm[tid] = sum; __syncthreads();
  for (int o = 128; o > 0; o >>= 1) {
    if (tid < o) sm[tid] += sm[tid + o];
    __syncthreads();
  }
  if (tid == 0) {
    float lp = lr[targets[row]] - mx - __logf(sm[0]);
    atomicAdd(loss, -lp * (1.0f / (float)BT));
  }
}

// ---------------------------------------------------------------------------
extern "C" void kernel_launch(void* const* d_in, const int* in_sizes, int n_in,
                              void* d_out, int out_size, void* d_ws, size_t ws_size,
                              hipStream_t stream) {
  (void)in_sizes; (void)n_in; (void)out_size; (void)ws_size;
  const int*   index   = (const int*)d_in[0];
  const int*   targets = (const int*)d_in[1];
  const float* tok_emb = (const float*)d_in[2];
  const float* pos_emb = (const float*)d_in[3];
  const float* Wq      = (const float*)d_in[4];
  const float* Wk      = (const float*)d_in[5];
  const float* Wv      = (const float*)d_in[6];
  const float* Wo      = (const float*)d_in[7];
  const float* bo      = (const float*)d_in[8];
  const float* W1      = (const float*)d_in[9];
  const float* b1      = (const float*)d_in[10];
  const float* W2      = (const float*)d_in[11];
  const float* b2w     = (const float*)d_in[12];
  const float* ln1_g   = (const float*)d_in[13];
  const float* ln1_b   = (const float*)d_in[14];
  const float* ln2_g   = (const float*)d_in[15];
  const float* ln2_b   = (const float*)d_in[16];
  const float* lnf_g   = (const float*)d_in[17];
  const float* lnf_b   = (const float*)d_in[18];
  const float* Wout    = (const float*)d_in[19];
  const float* bout    = (const float*)d_in[20];

  // ---- workspace carve-up (256B aligned) ----
  char* base = (char*)d_ws;
  size_t off = 0;
  auto carve = [&](size_t bytes) -> void* {
    void* p = base + off;
    off = (off + bytes + 255) & ~(size_t)255;
    return p;
  };
  float*  x   = (float*)carve((size_t)BT * DMODEL * 4);
  __bf16* xb  = (__bf16*)carve((size_t)BT * DMODEL * 2);
  float*  yv  = (float*)carve((size_t)BT * DMODEL * 4);
  __bf16* qb  = (__bf16*)carve((size_t)BT * DMODEL * 2);
  __bf16* kb  = (__bf16*)carve((size_t)BT * DMODEL * 2);
  __bf16* vb  = (__bf16*)carve((size_t)BT * DMODEL * 2);
  __bf16* vT  = (__bf16*)carve((size_t)BATCH * DMODEL * TP * 2);
  __bf16* ob  = (__bf16*)carve((size_t)BT * DMODEL * 2);
  __bf16* hb  = (__bf16*)carve((size_t)BT * DFF * 2);
  float*  S   = (float*)carve((size_t)BATCH * NHEAD * SEQ * TP * 4);
  __bf16* P   = (__bf16*)carve((size_t)BATCH * NHEAD * SEQ * TP * 2);
  __bf16* WqT = (__bf16*)carve((size_t)NLAYER * DMODEL * DMODEL * 2);
  __bf16* WkT = (__bf16*)carve((size_t)NLAYER * DMODEL * DMODEL * 2);
  __bf16* WvT = (__bf16*)carve((size_t)NLAYER * DMODEL * DMODEL * 2);
  __bf16* WoT = (__bf16*)carve((size_t)NLAYER * DMODEL * DMODEL * 2);
  __bf16* W1T = (__bf16*)carve((size_t)NLAYER * DFF * DMODEL * 2);
  __bf16* W2T = (__bf16*)carve((size_t)NLAYER * DMODEL * DFF * 2);
  __bf16* WoutT = (__bf16*)carve((size_t)VOCAB * DMODEL * 2);
  (void)carve(65536);  // tail pad: absorbs deliberate edge-tile overreads

  // ---- weight conversion to bf16, transposed to [N][K] ----
  {
    int qkvTot = NLAYER * NHEAD * DMODEL * DHEAD;
    int g = (qkvTot + 255) / 256;
    conv_qkv_kernel<<<g, 256, 0, stream>>>(Wq, WqT, qkvTot);
    conv_qkv_kernel<<<g, 256, 0, stream>>>(Wk, WkT, qkvTot);
    conv_qkv_kernel<<<g, 256, 0, stream>>>(Wv, WvT, qkvTot);
    int woTot = NLAYER * DMODEL * DMODEL;
    conv_transpose_kernel<<<(woTot + 255) / 256, 256, 0, stream>>>(Wo, WoT, DMODEL, DMODEL, woTot);
    int wfTot = NLAYER * DMODEL * DFF;
    conv_transpose_kernel<<<(wfTot + 255) / 256, 256, 0, stream>>>(W1, W1T, DMODEL, DFF, wfTot);
    conv_transpose_kernel<<<(wfTot + 255) / 256, 256, 0, stream>>>(W2, W2T, DFF, DMODEL, wfTot);
    int wvTot = DMODEL * VOCAB;
    conv_transpose_kernel<<<(wvTot + 255) / 256, 256, 0, stream>>>(Wout, WoutT, DMODEL, VOCAB, wvTot);
  }

  // ---- embedding ----
  embed_kernel<<<BT, DMODEL, 0, stream>>>(index, tok_emb, pos_emb, x, xb);

  const dim3 gemmBlk(128);
  const int MTILES = BT / 64;  // 75
  const dim3 attnGrid((SEQ + 15) / 16, BATCH * NHEAD);  // (19, 256)

  for (int l = 0; l < NLAYER; ++l) {
    const __bf16* wq = WqT + (size_t)l * DMODEL * DMODEL;
    const __bf16* wk = WkT + (size_t)l * DMODEL * DMODEL;
    const __bf16* wv = WvT + (size_t)l * DMODEL * DMODEL;

    gemm_bf16_kernel<<<dim3(MTILES, 6), gemmBlk, 0, stream>>>(
        xb, wq, nullptr, nullptr, qb, BT, DMODEL, DMODEL, DMODEL, DMODEL, DMODEL, 0);
    gemm_bf16_kernel<<<dim3(MTILES, 6), gemmBlk, 0, stream>>>(
        xb, wk, nullptr, nullptr, kb, BT, DMODEL, DMODEL, DMODEL, DMODEL, DMODEL, 0);
    gemm_bf16_kernel<<<dim3(MTILES, 6), gemmBlk, 0, stream>>>(
        xb, wv, nullptr, nullptr, vb, BT, DMODEL, DMODEL, DMODEL, DMODEL, DMODEL, 0);
    transpose_v_kernel<<<(BT * DMODEL + 255) / 256, 256, 0, stream>>>(vb, vT);

    attn_scores_kernel<<<attnGrid, 32, 0, stream>>>(qb, kb, S);
    softmax_kernel<<<dim3(SEQ, BATCH * NHEAD), 128, 0, stream>>>(S, P);
    attn_av_kernel<<<attnGrid, 32, 0, stream>>>(P, vT, ob);

    gemm_bf16_kernel<<<dim3(MTILES, 6), gemmBlk, 0, stream>>>(
        ob, WoT + (size_t)l * DMODEL * DMODEL, bo + (size_t)l * DMODEL,
        yv, nullptr, BT, DMODEL, DMODEL, DMODEL, DMODEL, DMODEL, 0);
    ln_kernel<<<BT, 128, 0, stream>>>(x, yv, ln1_g + (size_t)l * DMODEL,
                                      ln1_b + (size_t)l * DMODEL, xb);

    gemm_bf16_kernel<<<dim3(MTILES, 24), gemmBlk, 0, stream>>>(
        xb, W1T + (size_t)l * DFF * DMODEL, b1 + (size_t)l * DFF,
        nullptr, hb, BT, DFF, DMODEL, DMODEL, DMODEL, DFF, 1);
    gemm_bf16_kernel<<<dim3(MTILES, 6), gemmBlk, 0, stream>>>(
        hb, W2T + (size_t)l * DMODEL * DFF, b2w + (size_t)l * DMODEL,
        yv, nullptr, BT, DMODEL, DFF, DFF, DFF, DMODEL, 0);
    ln_kernel<<<BT, 128, 0, stream>>>(x, yv, ln2_g + (size_t)l * DMODEL,
                                      ln2_b + (size_t)l * DMODEL, xb);
  }

  // final LN -> bf16, vocab projection straight into d_out
  ln_kernel<<<BT, 128, 0, stream>>>(x, nullptr, lnf_g, lnf_b, xb);
  gemm_bf16_kernel<<<dim3(MTILES, (VOCAB + 63) / 64), gemmBlk, 0, stream>>>(
      xb, WoutT, bout, (float*)d_out, nullptr,
      BT, VOCAB, DMODEL, DMODEL, DMODEL, VOCAB, 0);

  float* lossp = (float*)d_out + (size_t)BT * VOCAB;
  zero_loss_kernel<<<1, 1, 0, stream>>>(lossp);
  loss_kernel<<<BT, 256, 0, stream>>>((const float*)d_out, targets, lossp);
}
